// PathPredictor_78658031058992
// MI455X (gfx1250) — compile-verified, matrix operand order checked by reading, not run on hardware
//
#include <hip/hip_runtime.h>
#include <hip/hip_bf16.h>

typedef __bf16 bf16_t;
typedef bf16_t v16bf __attribute__((ext_vector_type(16)));
typedef bf16_t v8bf  __attribute__((ext_vector_type(8)));
typedef float  v8f   __attribute__((ext_vector_type(8)));

namespace {
constexpr int kH   = 200;    // hidden
constexpr int kH3  = 600;    // 3*H (GRU gates)
constexpr int kR   = 500;    // relations
constexpr int kE   = 20000;  // entities
constexpr int kB   = 4096;   // batch
constexpr int kT   = 12;     // cur_ts (compile-time per problem spec)
constexpr int kPL  = 8;      // PATH_LEN
constexpr int kStartTs = kT - kPL; // 4
constexpr int kKP  = 224;    // 200 padded to mult of 32
constexpr int kKPR = 512;    // 500 padded to mult of 32
}

// ---------------------------------------------------------------------------
// WMMA fragment loader: bf16 16x32 A-tile (or Bt-tile) per ISA 7.12.2 layout.
// lane<16  : row = lane,    K = [0..7]  and [16..23]
// lane>=16 : row = lane-16, K = [8..15] and [24..31]
// p points at (row, kbase) of the tile; two 16B loads, gap of 16 elements.
// ---------------------------------------------------------------------------
__device__ __forceinline__ v16bf load_frag32(const bf16_t* __restrict__ p) {
  v8bf lo = *reinterpret_cast<const v8bf*>(p);
  v8bf hi = *reinterpret_cast<const v8bf*>(p + 16);
  return __builtin_shufflevector(lo, hi, 0,1,2,3,4,5,6,7,8,9,10,11,12,13,14,15);
}

// ---------------------------------------------------------------------------
// Generic bf16 WMMA GEMM:  C[M,N] = A[M,Kp] * Bt[N,Kp]^T (+ bias[N])
// Block = 256 threads = 8 waves. Wave w -> rows [bx*128 + 16w, +16),
// cols [by*64, +64) as 4 accumulators sharing one A fragment.
// ---------------------------------------------------------------------------
__global__ __launch_bounds__(256) void k_wmma_gemm(
    const bf16_t* __restrict__ A, int lda,
    const bf16_t* __restrict__ Bt, int ldb,
    const float* __restrict__ bias,
    float* __restrict__ Cf, int ldcf,
    bf16_t* __restrict__ Cb, int ldcb,
    int M, int N, int Kp)
{
  const int wave = threadIdx.x >> 5;
  const int lane = threadIdx.x & 31;
  const int m0 = blockIdx.x * 128 + wave * 16;
  const int n0 = blockIdx.y * 64;
  const int frow = lane & 15;
  const int kb = (lane >> 4) << 3;

  int am = m0 + frow; if (am > M - 1) am = M - 1;      // clamp OOB reads
  const bf16_t* ap = A + (size_t)am * lda + kb;

  const bf16_t* bp0; const bf16_t* bp1; const bf16_t* bp2; const bf16_t* bp3;
  {
    int n;
    n = n0 + frow;      if (n > N - 1) n = 0; bp0 = Bt + (size_t)n * ldb + kb;
    n = n0 + 16 + frow; if (n > N - 1) n = 0; bp1 = Bt + (size_t)n * ldb + kb;
    n = n0 + 32 + frow; if (n > N - 1) n = 0; bp2 = Bt + (size_t)n * ldb + kb;
    n = n0 + 48 + frow; if (n > N - 1) n = 0; bp3 = Bt + (size_t)n * ldb + kb;
  }

  v8f a0 = {}, a1 = {}, a2 = {}, a3 = {};
  for (int k = 0; k < Kp; k += 32) {
    v16bf af = load_frag32(ap + k);
    v16bf b0 = load_frag32(bp0 + k);
    v16bf b1 = load_frag32(bp1 + k);
    v16bf b2 = load_frag32(bp2 + k);
    v16bf b3 = load_frag32(bp3 + k);
    a0 = __builtin_amdgcn_wmma_f32_16x16x32_bf16(false, af, false, b0, (short)0, a0, false, false);
    a1 = __builtin_amdgcn_wmma_f32_16x16x32_bf16(false, af, false, b1, (short)0, a1, false, false);
    a2 = __builtin_amdgcn_wmma_f32_16x16x32_bf16(false, af, false, b2, (short)0, a2, false, false);
    a3 = __builtin_amdgcn_wmma_f32_16x16x32_bf16(false, af, false, b3, (short)0, a3, false, false);
  }

  // C layout: acc element r -> m = m0 + r + 8*(lane>=16), n = n0 + 16f + (lane&15)
  const int nloc = lane & 15;
  const int moff = (lane >> 4) << 3;
  v8f accs[4] = {a0, a1, a2, a3};
#pragma unroll
  for (int f = 0; f < 4; ++f) {
    int n = n0 + 16 * f + nloc;
    if (n >= N) continue;
    float bv = bias ? bias[n] : 0.0f;
#pragma unroll
    for (int r = 0; r < 8; ++r) {
      int m = m0 + moff + r;
      if (m >= M) continue;
      float v = accs[f][r] + bv;
      if (Cf) Cf[(size_t)m * ldcf + n] = v;
      if (Cb) Cb[(size_t)m * ldcb + n] = (bf16_t)v;
    }
  }
}

// ---------------------------------------------------------------------------
// Final scoring GEMM: score = sigmoid(M_b * P_b^T); fused one-hot BCE partial
// sums reduced in LDS and atomically accumulated into loss_acc.
// ---------------------------------------------------------------------------
__global__ __launch_bounds__(256) void k_wmma_score(
    const bf16_t* __restrict__ A,   // kB x kKP
    const bf16_t* __restrict__ Bt,  // kE x kKP
    const int* __restrict__ triples,
    float* __restrict__ score,      // kB x kE  (d_out + 2)
    float* __restrict__ loss_acc,   // d_out + 1
    int M, int N, int Kp)
{
  const int wave = threadIdx.x >> 5;
  const int lane = threadIdx.x & 31;
  const int m0 = blockIdx.x * 128 + wave * 16;
  const int n0 = blockIdx.y * 64;
  const int frow = lane & 15;
  const int kb = (lane >> 4) << 3;

  int am = m0 + frow; if (am > M - 1) am = M - 1;
  const bf16_t* ap = A + (size_t)am * Kp + kb;
  const bf16_t* bp0; const bf16_t* bp1; const bf16_t* bp2; const bf16_t* bp3;
  {
    int n;
    n = n0 + frow;      if (n > N - 1) n = 0; bp0 = Bt + (size_t)n * Kp + kb;
    n = n0 + 16 + frow; if (n > N - 1) n = 0; bp1 = Bt + (size_t)n * Kp + kb;
    n = n0 + 32 + frow; if (n > N - 1) n = 0; bp2 = Bt + (size_t)n * Kp + kb;
    n = n0 + 48 + frow; if (n > N - 1) n = 0; bp3 = Bt + (size_t)n * Kp + kb;
  }

  v8f a0 = {}, a1 = {}, a2 = {}, a3 = {};
  for (int k = 0; k < Kp; k += 32) {
    v16bf af = load_frag32(ap + k);
    v16bf b0 = load_frag32(bp0 + k);
    v16bf b1 = load_frag32(bp1 + k);
    v16bf b2 = load_frag32(bp2 + k);
    v16bf b3 = load_frag32(bp3 + k);
    a0 = __builtin_amdgcn_wmma_f32_16x16x32_bf16(false, af, false, b0, (short)0, a0, false, false);
    a1 = __builtin_amdgcn_wmma_f32_16x16x32_bf16(false, af, false, b1, (short)0, a1, false, false);
    a2 = __builtin_amdgcn_wmma_f32_16x16x32_bf16(false, af, false, b2, (short)0, a2, false, false);
    a3 = __builtin_amdgcn_wmma_f32_16x16x32_bf16(false, af, false, b3, (short)0, a3, false, false);
  }

  const int nloc = lane & 15;
  const int moff = (lane >> 4) << 3;
  v8f accs[4] = {a0, a1, a2, a3};
  float lsum = 0.0f;
#pragma unroll
  for (int f = 0; f < 4; ++f) {
    int n = n0 + 16 * f + nloc;
    if (n >= N) continue;
#pragma unroll
    for (int r = 0; r < 8; ++r) {
      int m = m0 + moff + r;
      if (m >= M) continue;
      float s = 1.0f / (1.0f + __expf(-accs[f][r]));
      score[(size_t)m * N + n] = s;
      float p = s;
      if (p < 1e-12f) p = 1e-12f;
      if (p > 1.0f - 1e-12f) p = 1.0f - 1e-12f;
      int obj = triples[m * 3 + 2];
      lsum += (n == obj) ? -__logf(p) : -__logf(1.0f - p);
    }
  }

  __shared__ float red[256];
  red[threadIdx.x] = lsum;
  __syncthreads();
  for (int o = 128; o > 0; o >>= 1) {
    if (threadIdx.x < (unsigned)o) red[threadIdx.x] += red[threadIdx.x + o];
    __syncthreads();
  }
  if (threadIdx.x == 0) atomicAdd(loss_acc, red[0]);
}

// ---------------------------------------------------------------------------
// Small fp32 GEMM (naive): Y[M,N] = X[M,K] * W[N,K]^T + b  (only 500-row work)
// ---------------------------------------------------------------------------
__global__ void k_lin_naive(const float* __restrict__ X, const float* __restrict__ W,
                            const float* __restrict__ bias, float* __restrict__ Y,
                            int M, int N, int K)
{
  int idx = blockIdx.x * blockDim.x + threadIdx.x;
  if (idx >= M * N) return;
  int m = idx / N, n = idx - m * N;
  float s = bias ? bias[n] : 0.0f;
  const float* x = X + (size_t)m * K;
  const float* w = W + (size_t)n * K;
  for (int k = 0; k < K; ++k) s += x[k] * w[k];
  Y[idx] = s;
}

// f32 (rows x K) -> bf16 (rows x Kp), zero-padded columns
__global__ void k_cvt_pad(const float* __restrict__ src, bf16_t* __restrict__ dst,
                          int rows, int K, int Kp)
{
  int idx = blockIdx.x * blockDim.x + threadIdx.x;
  if (idx >= rows * Kp) return;
  int r = idx / Kp, k = idx - r * Kp;
  dst[idx] = (k < K) ? (bf16_t)src[(size_t)r * K + k] : (bf16_t)0.0f;
}

// mapped_rel (kR x kH) -> transpose bf16 (kH x kKPR), zero-padded
__global__ void k_cvt_mappedT(const float* __restrict__ mapped, bf16_t* __restrict__ dst)
{
  int idx = blockIdx.x * blockDim.x + threadIdx.x;
  if (idx >= kH * kKPR) return;
  int h = idx / kKPR, r = idx - h * kKPR;
  dst[idx] = (r < kR) ? (bf16_t)mapped[(size_t)r * kH + h] : (bf16_t)0.0f;
}

// q_rel gather: f32 + padded bf16
__global__ void k_gather_qrel(const float* __restrict__ mapped, const int* __restrict__ triples,
                              float* __restrict__ qf, bf16_t* __restrict__ qb)
{
  int idx = blockIdx.x * blockDim.x + threadIdx.x;
  if (idx >= kB * kKP) return;
  int b = idx / kKP, k = idx - b * kKP;
  int rel = triples[b * 3 + 1];
  float v = (k < kH) ? mapped[(size_t)rel * kH + k] : 0.0f;
  if (k < kH) qf[(size_t)b * kH + k] = v;
  qb[idx] = (bf16_t)v;
}

// masked softmax over relations for one timestep -> padded bf16 attention
__global__ __launch_bounds__(512) void k_softmax(
    const float* __restrict__ base, const float* __restrict__ partial, int ts,
    bf16_t* __restrict__ attn)
{
  __shared__ float red[512];
  const int b = blockIdx.x;
  const int r = threadIdx.x;
  const bool act = (r < kR);
  float val = -3.0e38f;
  if (act) {
    float a = partial[(size_t)b * (kT * kR) + (size_t)ts * kR + r] * base[(size_t)b * kR + r];
    val = (a == 0.0f) ? -1.0e9f : a;
  }
  red[r] = val; __syncthreads();
  for (int o = 256; o > 0; o >>= 1) { if (r < o) red[r] = fmaxf(red[r], red[r + o]); __syncthreads(); }
  float mx = red[0]; __syncthreads();
  float e = act ? __expf(val - mx) : 0.0f;
  red[r] = e; __syncthreads();
  for (int o = 256; o > 0; o >>= 1) { if (r < o) red[r] += red[r + o]; __syncthreads(); }
  float inv = 1.0f / red[0];
  attn[(size_t)b * kKPR + r] = act ? (bf16_t)(e * inv) : (bf16_t)0.0f;
}

// GRU gate combine: h' = (1-z)*n + z*h  (also used for gru2 with hb = null)
__global__ void k_gru(const float* __restrict__ gi, const float* __restrict__ gh,
                      const float* hprev, float* hf, bf16_t* hb)
{
  int idx = blockIdx.x * blockDim.x + threadIdx.x;
  if (idx >= kB * kH) return;
  int b = idx / kH, j = idx - b * kH;
  const float* gib = gi + (size_t)b * kH3;
  const float* ghb = gh + (size_t)b * kH3;
  float rg = 1.0f / (1.0f + __expf(-(gib[j] + ghb[j])));
  float zg = 1.0f / (1.0f + __expf(-(gib[kH + j] + ghb[kH + j])));
  float ng = tanhf(gib[2 * kH + j] + rg * ghb[2 * kH + j]);
  float hv = (1.0f - zg) * ng + zg * hprev[idx];
  hf[idx] = hv;
  if (hb) hb[(size_t)b * kKP + j] = (bf16_t)hv;
}

// predicted_hist = 0.1*t3 + q_rel  (f32 + padded bf16)
__global__ void k_ph(const float* __restrict__ t3, const float* __restrict__ qrel,
                     float* __restrict__ phf, bf16_t* __restrict__ phb)
{
  int idx = blockIdx.x * blockDim.x + threadIdx.x;
  if (idx >= kB * kKP) return;
  int b = idx / kKP, k = idx - b * kKP;
  float v = (k < kH) ? (0.1f * t3[(size_t)b * kH + k] + qrel[(size_t)b * kH + k]) : 0.0f;
  if (k < kH) phf[(size_t)b * kH + k] = v;
  phb[idx] = (bf16_t)v;
}

// M_b[b,k] = P[sub[b],k] * gru2[b,k]  (padded bf16)
__global__ void k_make_M(const float* __restrict__ Pf, const float* __restrict__ gru2,
                         const int* __restrict__ triples, bf16_t* __restrict__ Mb)
{
  int idx = blockIdx.x * blockDim.x + threadIdx.x;
  if (idx >= kB * kKP) return;
  int b = idx / kKP, k = idx - b * kKP;
  int sub = triples[b * 3 + 0];
  float v = (k < kH) ? Pf[(size_t)sub * kH + k] * gru2[(size_t)b * kH + k] : 0.0f;
  Mb[idx] = (bf16_t)v;
}

// match_loss partial sums
__global__ __launch_bounds__(256) void k_match_loss(const float* __restrict__ ph,
                                                    const float* __restrict__ h,
                                                    float* __restrict__ acc)
{
  __shared__ float red[256];
  float s = 0.0f;
  for (size_t i = (size_t)blockIdx.x * blockDim.x + threadIdx.x;
       i < (size_t)kB * kH; i += (size_t)gridDim.x * blockDim.x) {
    float d = ph[i] - h[i];
    s += d * d;
  }
  red[threadIdx.x] = s; __syncthreads();
  for (int o = 128; o > 0; o >>= 1) {
    if (threadIdx.x < (unsigned)o) red[threadIdx.x] += red[threadIdx.x + o];
    __syncthreads();
  }
  if (threadIdx.x == 0) atomicAdd(acc, red[0]);
}

__global__ void k_zero32(unsigned int* __restrict__ p, size_t n)
{
  for (size_t i = (size_t)blockIdx.x * blockDim.x + threadIdx.x;
       i < n; i += (size_t)gridDim.x * blockDim.x) p[i] = 0u;
}

__global__ void k_finalize(float* out01)
{
  if (threadIdx.x == 0) {
    out01[0] *= 1.0f / ((float)kB * (float)kH);
    out01[1] *= 1.0f / ((float)kB * (float)kE);
  }
}

// ---------------------------------------------------------------------------
static void gemm(hipStream_t s, const bf16_t* A, int lda, const bf16_t* Bt, int ldb,
                 const float* bias, float* Cf, int ldcf, bf16_t* Cb, int ldcb,
                 int M, int N, int Kp)
{
  dim3 grid((M + 127) / 128, (N + 63) / 64);
  k_wmma_gemm<<<grid, 256, 0, s>>>(A, lda, Bt, ldb, bias, Cf, ldcf, Cb, ldcb, M, N, Kp);
}

static inline dim3 g1(long long n) { return dim3((unsigned)((n + 255) / 256)); }

extern "C" void kernel_launch(void* const* d_in, const int* in_sizes, int n_in,
                              void* d_out, int out_size, void* d_ws, size_t ws_size,
                              hipStream_t stream)
{
  (void)in_sizes; (void)n_in; (void)out_size; (void)ws_size;

  const float* pre_emb = (const float*)d_in[0];
  const float* r_emb   = (const float*)d_in[1];
  const float* partial = (const float*)d_in[2];
  /* d_in[3] = cur_ts, known at compile time (kT) */
  const float* Wm1 = (const float*)d_in[4];
  const float* bm1 = (const float*)d_in[5];
  const float* Wm2 = (const float*)d_in[6];
  const float* bm2 = (const float*)d_in[7];
  const float* Wh1 = (const float*)d_in[8];
  const float* bh1 = (const float*)d_in[9];
  const float* Wh2 = (const float*)d_in[10];
  const float* bh2 = (const float*)d_in[11];
  const float* Wattn  = (const float*)d_in[12];
  const float* battn  = (const float*)d_in[13];
  const float* Walign = (const float*)d_in[14];
  const float* balign = (const float*)d_in[15];
  const float* W_ih = (const float*)d_in[16];
  const float* b_ih = (const float*)d_in[17];
  const float* W_hh = (const float*)d_in[18];
  const float* b_hh = (const float*)d_in[19];
  const int* triples = (const int*)d_in[20];

  float* out = (float*)d_out;         // [match_loss, path_loss, score(kB*kE)]
  float* score = out + 2;

  // ---- workspace bump allocator (256B aligned) ----
  char* wp = (char*)d_ws;
  auto take = [&](size_t bytes) -> void* {
    void* p = (void*)wp;
    wp += (bytes + 255) & ~(size_t)255;
    return p;
  };

  // bf16 staging region (zeroed every call so pad columns are always 0)
  char* bf_start = wp;
  bf16_t* mapped_b  = (bf16_t*)take((size_t)kR * kKP * 2);
  bf16_t* mappedT_b = (bf16_t*)take((size_t)kH * kKPR * 2);
  bf16_t* qrel_b    = (bf16_t*)take((size_t)kB * kKP * 2);
  bf16_t* attnq_b   = (bf16_t*)take((size_t)kB * kKP * 2);
  bf16_t* attn_b    = (bf16_t*)take((size_t)kB * kKPR * 2);
  bf16_t* x_b       = (bf16_t*)take((size_t)kB * kKP * 2);
  bf16_t* t2_b      = (bf16_t*)take((size_t)kB * kKP * 2);
  bf16_t* h_b       = (bf16_t*)take((size_t)kB * kKP * 2);
  bf16_t* ph_b      = (bf16_t*)take((size_t)kB * kKP * 2);
  bf16_t* M_b       = (bf16_t*)take((size_t)kB * kKP * 2);
  bf16_t* pre_b     = (bf16_t*)take((size_t)kE * kKP * 2);
  bf16_t* P_b       = (bf16_t*)take((size_t)kE * kKP * 2);
  bf16_t* Wattn_b   = (bf16_t*)take((size_t)kH * kKP * 2);
  bf16_t* Walign_b  = (bf16_t*)take((size_t)kH * kKP * 2);
  bf16_t* Wh1_b     = (bf16_t*)take((size_t)kH * kKP * 2);
  bf16_t* Wh2_b     = (bf16_t*)take((size_t)kH * kKP * 2);
  bf16_t* Wih_b     = (bf16_t*)take((size_t)kH3 * kKP * 2);
  bf16_t* Whh_b     = (bf16_t*)take((size_t)kH3 * kKP * 2);
  size_t bf_bytes = (size_t)(wp - bf_start);

  // fp32 buffers
  float* tmp1_f   = (float*)take((size_t)kR * 400 * 4);
  float* mapped_f = (float*)take((size_t)kR * kH * 4);
  float* qrel_f   = (float*)take((size_t)kB * kH * 4);
  float* base_f   = (float*)take((size_t)kB * kR * 4);
  float* gi_f     = (float*)take((size_t)kB * kH3 * 4);
  float* gh_f     = (float*)take((size_t)kB * kH3 * 4);
  float* h_f      = (float*)take((size_t)kB * kH * 4);
  float* t3_f     = (float*)take((size_t)kB * kH * 4);
  float* ph_f     = (float*)take((size_t)kB * kH * 4);
  float* gru2_f   = (float*)take((size_t)kB * kH * 4);
  float* P_f      = (float*)take((size_t)kE * kH * 4);

  // ---- init: zero bf16 region, h state, loss accumulators ----
  k_zero32<<<2048, 256, 0, stream>>>((unsigned int*)bf_start, bf_bytes / 4);
  k_zero32<<<512, 256, 0, stream>>>((unsigned int*)h_f, (size_t)kB * kH);
  k_zero32<<<1, 32, 0, stream>>>((unsigned int*)out, 2);

  // ---- mapped_rel = (r_emb @ Wm1^T + bm1) @ Wm2^T + bm2 (small, fp32) ----
  k_lin_naive<<<g1((long long)kR * 400), 256, 0, stream>>>(r_emb, Wm1, bm1, tmp1_f, kR, 400, kH);
  k_lin_naive<<<g1((long long)kR * kH), 256, 0, stream>>>(tmp1_f, Wm2, bm2, mapped_f, kR, kH, 400);
  k_cvt_pad<<<g1((long long)kR * kKP), 256, 0, stream>>>(mapped_f, mapped_b, kR, kH, kKP);
  k_cvt_mappedT<<<g1((long long)kH * kKPR), 256, 0, stream>>>(mapped_f, mappedT_b);

  // ---- weight + embedding conversions to padded bf16 ----
  k_cvt_pad<<<g1((long long)kH * kKP), 256, 0, stream>>>(Wattn, Wattn_b, kH, kH, kKP);
  k_cvt_pad<<<g1((long long)kH * kKP), 256, 0, stream>>>(Walign, Walign_b, kH, kH, kKP);
  k_cvt_pad<<<g1((long long)kH * kKP), 256, 0, stream>>>(Wh1, Wh1_b, kH, kH, kKP);
  k_cvt_pad<<<g1((long long)kH * kKP), 256, 0, stream>>>(Wh2, Wh2_b, kH, kH, kKP);
  k_cvt_pad<<<g1((long long)kH3 * kKP), 256, 0, stream>>>(W_ih, Wih_b, kH3, kH, kKP);
  k_cvt_pad<<<g1((long long)kH3 * kKP), 256, 0, stream>>>(W_hh, Whh_b, kH3, kH, kKP);
  k_cvt_pad<<<g1((long long)kE * kKP), 256, 0, stream>>>(pre_emb, pre_b, kE, kH, kKP);

  // ---- q_rel gather ----
  k_gather_qrel<<<g1((long long)kB * kKP), 256, 0, stream>>>(mapped_f, triples, qrel_f, qrel_b);

  // ---- base_scores = (q_rel @ Wattn^T + battn) @ mapped_rel^T ----
  gemm(stream, qrel_b, kKP, Wattn_b, kKP, battn, nullptr, 0, attnq_b, kKP, kB, kH, kKP);
  gemm(stream, attnq_b, kKP, mapped_b, kKP, nullptr, base_f, kR, nullptr, 0, kB, kR, kKP);

  // ---- GRU scan over PATH_LEN steps ----
  for (int t = 0; t < kPL; ++t) {
    k_softmax<<<kB, 512, 0, stream>>>(base_f, partial, kStartTs + t, attn_b);
    // x = attn @ mapped_rel   (Bt = mapped_rel^T, N x K = kH x kKPR)
    gemm(stream, attn_b, kKPR, mappedT_b, kKPR, nullptr, nullptr, 0, x_b, kKP, kB, kH, kKPR);
    gemm(stream, x_b, kKP, Wih_b, kKP, b_ih, gi_f, kH3, nullptr, 0, kB, kH3, kKP);
    gemm(stream, h_b, kKP, Whh_b, kKP, b_hh, gh_f, kH3, nullptr, 0, kB, kH3, kKP);
    k_gru<<<g1((long long)kB * kH), 256, 0, stream>>>(gi_f, gh_f, h_f, h_f, h_b);
  }

  // ---- predicted_hist = 0.1*((q_rel@Wh1^T+bh1)@Wh2^T+bh2) + q_rel ----
  gemm(stream, qrel_b, kKP, Wh1_b, kKP, bh1, nullptr, 0, t2_b, kKP, kB, kH, kKP);
  gemm(stream, t2_b, kKP, Wh2_b, kKP, bh2, t3_f, kH, nullptr, 0, kB, kH, kKP);
  k_ph<<<g1((long long)kB * kKP), 256, 0, stream>>>(t3_f, qrel_f, ph_f, ph_b);

  // ---- match_loss = mean((ph - h)^2) ----
  k_match_loss<<<512, 256, 0, stream>>>(ph_f, h_f, out + 0);

  // ---- gru2 = gru_cell(q_rel, predicted_hist) ----
  gemm(stream, qrel_b, kKP, Wih_b, kKP, b_ih, gi_f, kH3, nullptr, 0, kB, kH3, kKP);
  gemm(stream, ph_b, kKP, Whh_b, kKP, b_hh, gh_f, kH3, nullptr, 0, kB, kH3, kKP);
  k_gru<<<g1((long long)kB * kH), 256, 0, stream>>>(gi_f, gh_f, ph_f, gru2_f, nullptr);

  // ---- P = pre_emb @ Walign^T + balign  (f32 for gather, bf16 for GEMM) ----
  gemm(stream, pre_b, kKP, Walign_b, kKP, balign, P_f, kH, P_b, kKP, kE, kH, kKP);

  // ---- M = P[sub] * gru2 ----
  k_make_M<<<g1((long long)kB * kKP), 256, 0, stream>>>(P_f, gru2_f, triples, M_b);

  // ---- score = sigmoid(M @ P^T), fused BCE path_loss ----
  {
    dim3 grid((kB + 127) / 128, (kE + 63) / 64);
    k_wmma_score<<<grid, 256, 0, stream>>>(M_b, P_b, triples, score, out + 1, kB, kE, kKP);
  }

  // ---- scale losses to means ----
  k_finalize<<<1, 32, 0, stream>>>(out);
}
